// GeodesicAwareAttention_9191230013582
// MI455X (gfx1250) — matrix operand order repaired and checked by reading, not compiled, not run on hardware
//
#include <hip/hip_runtime.h>
#include <math.h>

// ---------------- problem constants ----------------
constexpr int B_  = 2;
constexpr int C_  = 256;
constexpr int N_  = 2304;   // 48*48
constexpr int M_  = 16;     // anchors
constexpr int NH_ = 4;
constexpr int HD_ = 64;
constexpr int KT_ = N_ / 32;          // 72 key tiles of 32
constexpr int QG_ = N_ / 64;          // 36 query groups of 64 (4 waves x 16)

// LDS row pitches (halves) chosen to stagger banks; byte pitch is 16B-multiple
constexpr int KP_ = 72;               // K/V tile pitch (144 B)
constexpr int PP_ = 24;               // phi tile pitch (48 B)

typedef _Float16 v16h __attribute__((ext_vector_type(16)));
typedef float    v8f  __attribute__((ext_vector_type(8)));

#if __has_builtin(__builtin_amdgcn_global_load_async_to_lds_b128)
#define HAVE_ASYNC_LDS 1
#else
#define HAVE_ASYNC_LDS 0
#endif

// pointer types matching the async-builtin signature:
// param0: AS1 (global) pointer to 16B int vector; param1: AS3 (LDS) pointer
typedef int gv4i __attribute__((vector_size(16)));
typedef __attribute__((address_space(1))) gv4i* as1_v4i_p;
typedef __attribute__((address_space(3))) gv4i* as3_v4i_p;

__device__ inline v8f wmma_f16(v16h a, v16h b, v8f c) {
  return __builtin_amdgcn_wmma_f32_16x16x32_f16(false, a, false, b, (short)0, c,
                                                false, false);
}

__device__ inline v8f zero8() {
  v8f z;
#pragma unroll
  for (int i = 0; i < 8; ++i) z[i] = 0.0f;
  return z;
}

// 16-byte global -> LDS chunk copy (async DMA when available)
__device__ inline void cp16(_Float16* dst, const _Float16* src) {
#if HAVE_ASYNC_LDS
  __builtin_amdgcn_global_load_async_to_lds_b128(
      (as1_v4i_p)(const void*)src, (as3_v4i_p)(void*)dst, 0, 0);
#else
  *(uint4*)dst = *(const uint4*)src;
#endif
}

__device__ inline void wait_async_lds() {
#if HAVE_ASYNC_LDS
#if __has_builtin(__builtin_amdgcn_s_wait_asynccnt)
  __builtin_amdgcn_s_wait_asynccnt(0);
#else
  asm volatile("s_wait_asynccnt 0" ::: "memory");
#endif
#endif
}

// ---- A-matrix (16x32 f16) load, row-major source, leading dim ld ----
// contiguous k order (B loads match): lane m = lane&15, half = lane>>4;
//   VGPR j : K = 2j + 16*half
__device__ inline v16h load_a16(const _Float16* p, int ld, int k0, int lane) {
  int m = lane & 15, hf = lane >> 4;
  const _Float16* row = p + (size_t)m * ld + k0;
  v16h a;
#pragma unroll
  for (int j = 0; j < 8; ++j) {
    int kk = 2 * j + 16 * hf;
    a[2 * j]     = row[kk];
    a[2 * j + 1] = row[kk + 1];
  }
  return a;
}

// ---- B-matrix (32x16 f16) from "transposed" source: B[k][n] = src[n][k] ----
// B layout: col n = lane&15, half = lane>>4; VGPR j : K = 2j + 16*half
__device__ inline v16h load_bT(const _Float16* p, int ld, int k0, int lane) {
  int n = lane & 15, hf = lane >> 4;
  const _Float16* row = p + (size_t)n * ld + k0;
  v16h b;
#pragma unroll
  for (int j = 0; j < 8; ++j) {
    int kk = 2 * j + 16 * hf;
    b[2 * j]     = row[kk];
    b[2 * j + 1] = row[kk + 1];
  }
  return b;
}

// ---- B-matrix from row-major [K x ncols] source: B[k][n] = p[k*ld + n] ----
__device__ inline v16h load_bRM(const _Float16* p, int ld, int lane) {
  int n = lane & 15, hf = lane >> 4;
  v16h b;
#pragma unroll
  for (int j = 0; j < 8; ++j) {
    int kk = 2 * j + 16 * hf;
    b[2 * j]     = p[(size_t)kk * ld + n];
    b[2 * j + 1] = p[(size_t)(kk + 1) * ld + n];
  }
  return b;
}

// ---- phi operands: real K = 16 anchors, zero-padded to K = 32 ----
__device__ inline v16h load_a_phi(const _Float16* p, int ld, int lane) {
  int m = lane & 15, hf = lane >> 4;
  const _Float16* row = p + (size_t)m * ld;
  v16h a;
#pragma unroll
  for (int i = 0; i < 16; ++i) a[i] = (_Float16)0.0f;
#pragma unroll
  for (int j = 0; j < 8 * (1 - hf); ++j) { // valid K: 2j in [0,16), hf == 0 only
    int kk = 2 * j;
    a[2 * j]     = row[kk];
    a[2 * j + 1] = row[kk + 1];
  }
  return a;
}

__device__ inline v16h load_b_phi(const _Float16* p, int ld, int lane) {
  int n = lane & 15, hf = lane >> 4;
  v16h b;
#pragma unroll
  for (int i = 0; i < 16; ++i) b[i] = (_Float16)0.0f;
  if (hf == 0) {                          // K = 2j in [0,16): all valid
    const _Float16* row = p + (size_t)n * ld;
#pragma unroll
    for (int j = 0; j < 8; ++j) {
      b[2 * j]     = row[2 * j];
      b[2 * j + 1] = row[2 * j + 1];
    }
  }
  return b;
}

// =====================================================================
// K0: weight f32 -> f16
// =====================================================================
__global__ void cvt_kernel(const float* wq, const float* wk, const float* wv,
                           const float* wo, _Float16* q, _Float16* k,
                           _Float16* v, _Float16* o) {
  int i = blockIdx.x * blockDim.x + threadIdx.x;
  if (i < C_ * C_) {
    q[i] = (_Float16)wq[i];
    k[i] = (_Float16)wk[i];
    v[i] = (_Float16)wv[i];
    o[i] = (_Float16)wo[i];
  }
}

// =====================================================================
// K1: geodesic PE MLP + residual; emits xpe f16, phi f16, phi_sq f32
// =====================================================================
__global__ __launch_bounds__(128) void pe_kernel(
    const float* __restrict__ x, const float* __restrict__ phi,
    const float* __restrict__ W1, const float* __restrict__ b1,
    const float* __restrict__ W2, const float* __restrict__ b2,
    _Float16* __restrict__ xpe, _Float16* __restrict__ phi16,
    float* __restrict__ phisq) {
  __shared__ float phs[M_];
  __shared__ float hid[128];
  int t = blockIdx.x;            // token in [0, B*N)
  int b = t / N_, nsp = t % N_;
  int tid = threadIdx.x;

  if (tid < M_) {
    float pv = phi[((size_t)b * M_ + tid) * N_ + nsp];
    phs[tid] = pv;
    phi16[(size_t)t * M_ + tid] = (_Float16)pv;
  }
  __syncthreads();

  if (tid == 0) {
    float s = 0.0f;
#pragma unroll
    for (int a = 0; a < M_; ++a) s += phs[a] * phs[a];
    phisq[t] = s;
  }

  {
    float h = b1[tid];
    const float* wr = W1 + (size_t)tid * M_;
#pragma unroll
    for (int a = 0; a < M_; ++a) h += phs[a] * wr[a];
    hid[tid] = 0.5f * h * (1.0f + erff(h * 0.70710678118654752f));
  }
  __syncthreads();

  for (int c = tid; c < C_; c += 128) {
    float pe = b2[c];
    const float* wr = W2 + (size_t)c * 128;
#pragma unroll 8
    for (int j = 0; j < 128; ++j) pe += hid[j] * wr[j];
    float xv = x[((size_t)b * C_ + c) * N_ + nsp];
    xpe[(size_t)t * C_ + c] = (_Float16)(xv + pe);
  }
}

// =====================================================================
// K2: Q/K/V = xpe @ W^T + b   (WMMA, one 16x16 tile per wave)
// =====================================================================
__global__ __launch_bounds__(256) void qkv_kernel(
    const _Float16* __restrict__ xpe, const _Float16* __restrict__ wq,
    const _Float16* __restrict__ wk, const _Float16* __restrict__ wv,
    const float* __restrict__ bq, const float* __restrict__ bk,
    const float* __restrict__ bv, _Float16* __restrict__ q,
    _Float16* __restrict__ k, _Float16* __restrict__ v) {
  constexpr int RT = (B_ * N_) / 16;    // 288 row tiles
  constexpr int CT = C_ / 16;           // 16 col tiles
  int wid = blockIdx.x * (blockDim.x >> 5) + (threadIdx.x >> 5);
  int mat = wid / (RT * CT);
  int rem = wid % (RT * CT);
  int rt = rem / CT, ct = rem % CT;
  int lane = threadIdx.x & 31;
  int n16 = lane & 15, hf = lane >> 4;

  const _Float16* W = (mat == 0) ? wq : (mat == 1) ? wk : wv;
  const float* bias = (mat == 0) ? bq : (mat == 1) ? bk : bv;
  _Float16* out = (mat == 0) ? q : (mat == 1) ? k : v;

  v8f acc;
  float bval = bias[ct * 16 + n16];
#pragma unroll
  for (int r = 0; r < 8; ++r) acc[r] = bval;

  const _Float16* arow = xpe + (size_t)(rt * 16) * C_;
  const _Float16* brow = W + (size_t)(ct * 16) * C_;
#pragma unroll
  for (int k0 = 0; k0 < C_; k0 += 32) {
    acc = wmma_f16(load_a16(arow, C_, k0, lane),
                   load_bT(brow, C_, k0, lane), acc);
  }

  _Float16* o = out + (size_t)(rt * 16) * C_ + ct * 16 + n16;
#pragma unroll
  for (int r = 0; r < 8; ++r) o[(size_t)(r + 8 * hf) * C_] = (_Float16)acc[r];
}

// =====================================================================
// K3: fused geodesic flash-attention.
// Block = 4 waves sharing one (b,h) and 64 queries; K/V/phi tiles staged
// to LDS once per block (async DMA + double buffer when available).
// =====================================================================
__global__ __launch_bounds__(128) void attn_kernel(
    const _Float16* __restrict__ q16, const _Float16* __restrict__ k16,
    const _Float16* __restrict__ v16, const _Float16* __restrict__ phi16,
    const float* __restrict__ phisq, const unsigned char* __restrict__ mask,
    const float* __restrict__ beta, const float* __restrict__ log_alpha,
    _Float16* __restrict__ o16) {
  __shared__ __align__(16) _Float16 kbuf[2][32 * KP_];
  __shared__ __align__(16) _Float16 vbuf[2][32 * KP_];
  __shared__ __align__(16) _Float16 phbuf[2][32 * PP_];
  __shared__ float    sbuf[4][16 * 32];
  __shared__ _Float16 pbuf[4][16 * 32];
  __shared__ float    rsc[4][16];
  __shared__ float    rinv[4][16];

  int tid = threadIdx.x;
  int w = tid >> 5;
  int lane = tid & 31;
  int n16 = lane & 15, hf = lane >> 4;

  int qg = blockIdx.x % QG_;
  int h  = (blockIdx.x / QG_) % NH_;
  int b  = blockIdx.x / (QG_ * NH_);
  int qt = qg * 4 + w;                      // this wave's 16-query tile

  const float scale = 0.125f;                                     // HD^-0.5
  const float coef  = -expf(log_alpha[0]) * 0.17677669529663687f; // 1/sqrt(2M)
  const float bh    = beta[h];

  const _Float16* kgb = k16 + (size_t)b * N_ * C_ + h * HD_;
  const _Float16* vgb = v16 + (size_t)b * N_ * C_ + h * HD_;
  const _Float16* pgb = phi16 + (size_t)b * N_ * M_;

  size_t tokq = (size_t)b * N_ + qt * 16;
  const _Float16* qbase = q16 + tokq * C_ + h * HD_;
  v16h aq0  = load_a16(qbase, C_, 0, lane);
  v16h aq1  = load_a16(qbase, C_, 32, lane);
  v16h aphi = load_a_phi(phi16 + tokq * M_, M_, lane);

  float psqq[8];
  float mq[8];
  {
    const float* pq = phisq + tokq;
    const unsigned char* mp = mask + tokq;
#pragma unroll
    for (int r = 0; r < 8; ++r) {
      int m = r + 8 * hf;
      psqq[r] = pq[m];
      mq[r] = mp[m] ? 1.0f : 0.0f;
    }
  }

  v8f acc0 = zero8(), acc1 = zero8(), acc2 = zero8(), acc3 = zero8();
  float mi = -1e30f, li = 0.0f;             // lanes 0..15 own rows

  // ---- block-cooperative tile staging: 16B chunks, coalesced ----
  auto stage = [&](int key0, int dbuf) {
    const _Float16* kg = kgb + (size_t)key0 * C_;
    const _Float16* vg = vgb + (size_t)key0 * C_;
    const _Float16* pg = pgb + (size_t)key0 * M_;
#pragma unroll
    for (int p = 0; p < 2; ++p) {
      int c = tid + p * 128;                // 256 chunks of 16B per tile
      int r = c >> 3, j = (c & 7) * 8;
      cp16(&kbuf[dbuf][r * KP_ + j], kg + (size_t)r * C_ + j);
      cp16(&vbuf[dbuf][r * KP_ + j], vg + (size_t)r * C_ + j);
    }
    if (tid < 64) {                         // 64 chunks for phi tile
      int r = tid >> 1, j = (tid & 1) * 8;
      cp16(&phbuf[dbuf][r * PP_ + j], pg + (size_t)r * M_ + j);
    }
  };

  stage(0, 0);

  for (int kt = 0; kt < KT_; ++kt) {
    int buf = kt & 1;
    int key0 = kt * 32;
    wait_async_lds();
    __syncthreads();                        // tile `buf` ready; prior reads done
    if (kt + 1 < KT_) stage((kt + 1) * 32, buf ^ 1);

    // ---- S = Q K^T + beta*log_k_geo over two 16-key halves ----
#pragma unroll
    for (int hc = 0; hc < 2; ++hc) {
      const _Float16* kb = &kbuf[buf][hc * 16 * KP_];
      v8f s = zero8();
      s = wmma_f16(aq0, load_bT(kb, KP_, 0, lane), s);
      s = wmma_f16(aq1, load_bT(kb, KP_, 32, lane), s);
      v16h bphi = load_b_phi(&phbuf[buf][hc * 16 * PP_], PP_, lane);
      v8f g = wmma_f16(aphi, bphi, zero8());

      int kk0 = key0 + hc * 16;
      float psk = phisq[(size_t)b * N_ + kk0 + n16];
      float mk = mask[(size_t)b * N_ + kk0 + n16] ? 1.0f : 0.0f;
#pragma unroll
      for (int r = 0; r < 8; ++r) {
        float pw = psqq[r] + psk - 2.0f * g[r];
        pw = pw < 0.0f ? 0.0f : pw;
        float sv = s[r] * scale + bh * coef * pw;
        if (mq[r] * mk == 0.0f) sv = -1e30f;
        int m = r + 8 * hf;
        sbuf[w][m * 32 + hc * 16 + n16] = sv;
      }
    }

    // ---- online softmax (wave-local; LDS ops are in-order per wave) ----
    if (lane < 16) {
      const float* srow = &sbuf[w][lane * 32];
      float tm = srow[0];
#pragma unroll
      for (int j = 1; j < 32; ++j) tm = fmaxf(tm, srow[j]);
      float nm = fmaxf(mi, tm);
      float f = __expf(mi - nm);
      float ssum = 0.0f;
      _Float16* prow = &pbuf[w][lane * 32];
#pragma unroll
      for (int j = 0; j < 32; ++j) {
        float e = __expf(srow[j] - nm);
        ssum += e;
        prow[j] = (_Float16)e;
      }
      li = li * f + ssum;
      mi = nm;
      rsc[w][lane] = f;
    }

    // ---- rescale running output, then O += P V ----
#pragma unroll
    for (int r = 0; r < 8; ++r) {
      float f = rsc[w][r + 8 * hf];
      acc0[r] *= f; acc1[r] *= f; acc2[r] *= f; acc3[r] *= f;
    }
    v16h ap = load_a16(&pbuf[w][0], 32, 0, lane);
    const _Float16* vb = &vbuf[buf][0];
    acc0 = wmma_f16(ap, load_bRM(vb + 0,  KP_, lane), acc0);
    acc1 = wmma_f16(ap, load_bRM(vb + 16, KP_, lane), acc1);
    acc2 = wmma_f16(ap, load_bRM(vb + 32, KP_, lane), acc2);
    acc3 = wmma_f16(ap, load_bRM(vb + 48, KP_, lane), acc3);
  }

  if (lane < 16) rinv[w][lane] = 1.0f / li;

  _Float16* ob = o16 + tokq * C_ + h * HD_ + n16;
#pragma unroll
  for (int r = 0; r < 8; ++r) {
    int m = r + 8 * hf;
    float inv = rinv[w][m];
    ob[(size_t)m * C_ + 0]  = (_Float16)(acc0[r] * inv);
    ob[(size_t)m * C_ + 16] = (_Float16)(acc1[r] * inv);
    ob[(size_t)m * C_ + 32] = (_Float16)(acc2[r] * inv);
    ob[(size_t)m * C_ + 48] = (_Float16)(acc3[r] * inv);
  }
}

// =====================================================================
// K4: out = attn @ Wo^T + bo, stored f32 transposed to [B,C,H,W]
// =====================================================================
__global__ __launch_bounds__(256) void oproj_kernel(
    const _Float16* __restrict__ attn, const _Float16* __restrict__ wo,
    const float* __restrict__ bo, float* __restrict__ out) {
  constexpr int CT = C_ / 16;
  int wid = blockIdx.x * (blockDim.x >> 5) + (threadIdx.x >> 5);
  int rt = wid / CT, ct = wid % CT;
  int lane = threadIdx.x & 31;
  int n16 = lane & 15, hf = lane >> 4;

  v8f acc;
  float bval = bo[ct * 16 + n16];
#pragma unroll
  for (int r = 0; r < 8; ++r) acc[r] = bval;

  const _Float16* arow = attn + (size_t)(rt * 16) * C_;
  const _Float16* brow = wo + (size_t)(ct * 16) * C_;
#pragma unroll
  for (int k0 = 0; k0 < C_; k0 += 32) {
    acc = wmma_f16(load_a16(arow, C_, k0, lane),
                   load_bT(brow, C_, k0, lane), acc);
  }

#pragma unroll
  for (int r = 0; r < 8; ++r) {
    int m = r + 8 * hf;
    int t = rt * 16 + m;
    int b = t / N_, nsp = t % N_;
    int c = ct * 16 + n16;
    out[((size_t)b * C_ + c) * N_ + nsp] = acc[r];
  }
}

// =====================================================================
extern "C" void kernel_launch(void* const* d_in, const int* in_sizes, int n_in,
                              void* d_out, int out_size, void* d_ws,
                              size_t ws_size, hipStream_t stream) {
  (void)in_sizes; (void)n_in; (void)out_size; (void)ws_size;
  const float* x   = (const float*)d_in[0];
  const float* phi = (const float*)d_in[1];
  const unsigned char* mask = (const unsigned char*)d_in[2];
  const float* Wq = (const float*)d_in[3];
  const float* bq = (const float*)d_in[4];
  const float* Wk = (const float*)d_in[5];
  const float* bk = (const float*)d_in[6];
  const float* Wv = (const float*)d_in[7];
  const float* bv = (const float*)d_in[8];
  const float* Wo = (const float*)d_in[9];
  const float* bo = (const float*)d_in[10];
  const float* W1 = (const float*)d_in[11];
  const float* b1 = (const float*)d_in[12];
  const float* W2 = (const float*)d_in[13];
  const float* b2 = (const float*)d_in[14];
  const float* log_alpha = (const float*)d_in[15];
  const float* beta = (const float*)d_in[16];

  char* ws = (char*)d_ws;
  size_t off = 0;
  auto take = [&](size_t bytes) -> char* {
    char* p = ws + off;
    off += (bytes + 255) & ~(size_t)255;
    return p;
  };
  _Float16* wq16 = (_Float16*)take((size_t)C_ * C_ * 2);
  _Float16* wk16 = (_Float16*)take((size_t)C_ * C_ * 2);
  _Float16* wv16 = (_Float16*)take((size_t)C_ * C_ * 2);
  _Float16* wo16 = (_Float16*)take((size_t)C_ * C_ * 2);
  _Float16* phi16 = (_Float16*)take((size_t)B_ * N_ * M_ * 2);
  float*    phisq = (float*)take((size_t)B_ * N_ * 4);
  _Float16* xpe16 = (_Float16*)take((size_t)B_ * N_ * C_ * 2);
  _Float16* q16   = (_Float16*)take((size_t)B_ * N_ * C_ * 2);
  _Float16* k16   = (_Float16*)take((size_t)B_ * N_ * C_ * 2);
  _Float16* v16   = (_Float16*)take((size_t)B_ * N_ * C_ * 2);
  _Float16* a16   = (_Float16*)take((size_t)B_ * N_ * C_ * 2);

  cvt_kernel<<<(C_ * C_ + 255) / 256, 256, 0, stream>>>(
      Wq, Wk, Wv, Wo, wq16, wk16, wv16, wo16);

  pe_kernel<<<B_ * N_, 128, 0, stream>>>(x, phi, W1, b1, W2, b2, xpe16, phi16,
                                         phisq);

  {
    int tiles = 3 * (B_ * N_ / 16) * (C_ / 16);   // 13824 waves
    qkv_kernel<<<tiles / 8, 256, 0, stream>>>(xpe16, wq16, wk16, wv16, bq, bk,
                                              bv, q16, k16, v16);
  }

  attn_kernel<<<B_ * NH_ * QG_, 128, 0, stream>>>(q16, k16, v16, phi16, phisq,
                                                  mask, beta, log_alpha, a16);

  {
    int tiles = (B_ * N_ / 16) * (C_ / 16);       // 4608 waves
    oproj_kernel<<<tiles / 8, 256, 0, stream>>>(a16, wo16, bo, (float*)d_out);
  }
}
